// DecoderOnlyMusicGenModel_41077067219516
// MI455X (gfx1250) — compile-verified, matrix operand order checked by reading, not run on hardware
//
#include <hip/hip_runtime.h>
#include <stdint.h>

// ---------------- model dimensions (compile-time) ----------------
constexpr int Dd = 512, Hh = 8, HDd = 64, Ll = 12, Ff = 2048, Vv = 2048,
              Rr = 1024, Cc = 64;
constexpr int Bb = 8, Tt = 1024, Mc = 128;
constexpr int BT = Bb * Tt;        // 8192 token rows
constexpr int NREL = 2 * Rr + 1;   // 2049 relative-position rows

typedef __bf16 v8bf  __attribute__((ext_vector_type(8)));
typedef __bf16 v16bf __attribute__((ext_vector_type(16)));
typedef float  v8f   __attribute__((ext_vector_type(8)));
typedef unsigned int u32x4 __attribute__((ext_vector_type(4)));
typedef int i32x4 __attribute__((ext_vector_type(4)));
typedef int i32x8 __attribute__((ext_vector_type(8)));

#define HAVE_TDM __has_builtin(__builtin_amdgcn_tensor_load_to_lds)

// ---------------- helpers ----------------
__device__ __forceinline__ __bf16 f2bf(float f) {
  unsigned u = __builtin_bit_cast(unsigned, f);
  u += 0x7FFFu + ((u >> 16) & 1u);                 // round-to-nearest-even
  unsigned short h = (unsigned short)(u >> 16);
  return __builtin_bit_cast(__bf16, h);
}

__device__ __forceinline__ v16bf join16(v8bf lo, v8bf hi) {
  v16bf r;
#pragma unroll
  for (int i = 0; i < 8; ++i) { r[i] = lo[i]; r[i + 8] = hi[i]; }
  return r;
}

// One WMMA 16x32 bf16 operand fragment.
// ISA layout (CDNA5 ISA): lane L<16 holds row M=L, K={kb..kb+7, kb+16..kb+23};
// lanes 16..31 hold the K={+8..+15, +24..+31} halves.  Works for both the A
// operand (rows = M) and the B operand when data is stored column(n)-major.
__device__ __forceinline__ v16bf load_frag(const __bf16* base, int row0,
                                           int rowStride, int kbase, int lane) {
  const int rr = lane & 15;
  const int kg = (lane >> 4) * 8;
  const __bf16* p = base + (size_t)(row0 + rr) * rowStride + kbase + kg;
  v8bf lo = *(const v8bf*)p;
  v8bf hi = *(const v8bf*)(p + 16);
  return join16(lo, hi);
}

__device__ __forceinline__ v8f wmma_bf16(v16bf a, v16bf b, v8f c) {
  return __builtin_amdgcn_wmma_f32_16x16x32_bf16(false, a, false, b,
                                                 (short)0, c, false, false);
}

__device__ __forceinline__ float redmax16(float v) {
#pragma unroll
  for (int m = 8; m >= 1; m >>= 1) v = fmaxf(v, __shfl_xor(v, m, 32));
  return v;
}
__device__ __forceinline__ float redsum16(float v) {
#pragma unroll
  for (int m = 8; m >= 1; m >>= 1) v += __shfl_xor(v, m, 32);
  return v;
}

#if HAVE_TDM
// Issue one Tensor-Data-Mover load of a 64x32 bf16 tile (row stride K elems)
// into LDS.  D# layout per CDNA5 ISA:
//   group0: count=1 | lds_addr | global_addr[56:0] | type=2
//   group1: data_size=2B, tensor_dim0=K, tensor_dim1=M, tile_dim0=32,
//           tile_dim1=64, tensor_dim0_stride=K
// groups 2/3 are zero (tensor rank <= 2).  Tracked by TENSORcnt.
__device__ __forceinline__ void tdm_load_tile_a(const __bf16* gsrc,
                                                unsigned lds_off, int K, int M) {
  const unsigned long long ga = (unsigned long long)(uintptr_t)gsrc;
  u32x4 g0;
  g0[0] = 1u;                                       // count=1, user descriptor
  g0[1] = lds_off;                                  // LDS byte address
  g0[2] = (unsigned)(ga & 0xFFFFFFFFu);             // global_addr[31:0]
  g0[3] = (unsigned)((ga >> 32) & 0x01FFFFFFu) | (2u << 30);  // [56:32]|type=2
  const unsigned uK = (unsigned)K, uM = (unsigned)M;
  i32x8 g1;
  g1[0] = (int)(1u << 16);                          // data_size = 2 bytes
  g1[1] = (int)((uK & 0xFFFFu) << 16);              // tensor_dim0 lo16
  g1[2] = (int)(((uK >> 16) & 0xFFFFu) | ((uM & 0xFFFFu) << 16));
  g1[3] = (int)(((uM >> 16) & 0xFFFFu) | (32u << 16));  // tile_dim0 = 32
  g1[4] = (int)64u;                                 // tile_dim1 = 64
  g1[5] = (int)uK;                                  // tensor_dim0_stride lo32
  g1[6] = 0;
  g1[7] = 0;
  const i32x4 z4 = {0, 0, 0, 0};
#if defined(__clang_major__) && __clang_major__ >= 23
  const i32x8 z8 = {0, 0, 0, 0, 0, 0, 0, 0};
  __builtin_amdgcn_tensor_load_to_lds(g0, g1, z4, z4, z8, 0);
#else
  __builtin_amdgcn_tensor_load_to_lds(g0, g1, z4, z4, 0);
#endif
}
#endif

// ---------------- generic WMMA GEMM:  C = A(bf16) x B(f32 weights) + bias ----
enum GemmMode { OUT_F32 = 0, OUT_BF16 = 1, OUT_BF16_RELU = 2, OUT_QKV = 3 };

template <int MODE>
__global__ __launch_bounds__(128) void gemm_kernel(
    const __bf16* __restrict__ A, const float* __restrict__ Bw,
    const float* __restrict__ bias, void* __restrict__ Cout,
    int M, int N, int K) {
  __shared__ __bf16 sA[64 * 32];   // [m][k]
  __shared__ __bf16 sB[64 * 40];   // [n][k], padded stride 40 (16B-aligned rows)

  const int tid  = threadIdx.x;
  const int lane = tid & 31;
  const int wave = tid >> 5;
  const int m0 = blockIdx.y * 64;
  const int n0 = blockIdx.x * 64;

  v8f acc[4];
#pragma unroll
  for (int s = 0; s < 4; ++s)
#pragma unroll
    for (int i = 0; i < 8; ++i) acc[s][i] = 0.0f;

  const int krow  = tid >> 2;          // 0..31
  const int nb    = (tid & 3) * 16;    // 0,16,32,48
#if HAVE_TDM
  const unsigned sA_off = (unsigned)(uintptr_t)sA;  // LDS byte address
#else
  const int arow  = tid >> 1;          // 0..63
  const int ahalf = (tid & 1) * 16;    // 0 or 16
#endif

  for (int k0 = 0; k0 < K; k0 += 32) {
#if HAVE_TDM
    // A tile (64x32 bf16) staged by the Tensor Data Mover; wave 0 issues the
    // DMA and drains TENSORcnt, the block barrier publishes it to all waves.
    if (wave == 0) {
      tdm_load_tile_a(A + (size_t)m0 * K + k0, sA_off, K, M);
      __builtin_amdgcn_s_wait_tensorcnt(0);
    }
#else
    const __bf16* as = A + (size_t)(m0 + arow) * K + k0 + ahalf;
    *(v8bf*)(sA + arow * 32 + ahalf)     = *(const v8bf*)as;
    *(v8bf*)(sA + arow * 32 + ahalf + 8) = *(const v8bf*)(as + 8);
#endif
    // stage B tile (f32 weights -> bf16), transposed to [n][k]
    const float* bs = Bw + (size_t)(k0 + krow) * N + n0 + nb;
#pragma unroll
    for (int j = 0; j < 16; ++j) sB[(nb + j) * 40 + krow] = f2bf(bs[j]);
    if (k0 + 32 < K)
      __builtin_prefetch(Bw + (size_t)(k0 + 32 + krow) * N + n0 + nb, 0, 1);
    __syncthreads();

    v16bf af = load_frag(sA, wave * 16, 32, 0, lane);
#pragma unroll
    for (int s = 0; s < 4; ++s) {
      v16bf bf = load_frag(sB, s * 16, 40, 0, lane);
      acc[s] = wmma_bf16(af, bf, acc[s]);
    }
    __syncthreads();
  }

  const int hi = lane >> 4;
  const int ln = lane & 15;
#pragma unroll
  for (int s = 0; s < 4; ++s) {
    const int n = n0 + s * 16 + ln;
    const float bv = bias[n];
#pragma unroll
    for (int r = 0; r < 8; ++r) {
      const int m = m0 + wave * 16 + r + 8 * hi;
      float v = acc[s][r] + bv;
      if (MODE == OUT_F32) {
        ((float*)Cout)[(size_t)m * N + n] = v;
      } else if (MODE == OUT_BF16) {
        ((__bf16*)Cout)[(size_t)m * N + n] = f2bf(v);
      } else if (MODE == OUT_BF16_RELU) {
        ((__bf16*)Cout)[(size_t)m * N + n] = f2bf(fmaxf(v, 0.0f));
      } else {  // OUT_QKV: row m = b*T + t, col n = h*HD + hd -> [B,H,T,HD]
        const int b = m >> 10, t = m & 1023;
        const int h = n >> 6, hd = n & 63;
        ((__bf16*)Cout)[(((size_t)b * Hh + h) * Tt + t) * HDd + hd] = f2bf(v);
      }
    }
  }
}

// ---------------- flash attention with banded relative positions ------------
__global__ __launch_bounds__(128) void attention_kernel(
    const __bf16* __restrict__ Q, const __bf16* __restrict__ Km,
    const __bf16* __restrict__ Vm, const __bf16* __restrict__ REL,
    const int* __restrict__ x, __bf16* __restrict__ O) {
  __shared__ float  relLDS[4][16 * 32];   // per-wave banded rel product
  __shared__ __bf16 pLDS[4][16 * 16];     // per-wave P tile (transpose hop)
  __shared__ __bf16 vLDS[4][64 * 32];     // per-wave V tile [hd][key], keys 16..31 = 0

  const int tid  = threadIdx.x;
  const int lane = tid & 31;
  const int wave = tid >> 5;
  const int hi = lane >> 4;
  const int ln = lane & 15;
  const int bh = blockIdx.y;
  const int b  = bh >> 3;   // H = 8
  const int h  = bh & 7;
  const int q0 = (blockIdx.x * 4 + wave) * 16;

  float*  relw = relLDS[wave];
  __bf16* pw   = pLDS[wave];
  __bf16* vw   = vLDS[wave];

  // zero the upper-16-key half of the V tile once (never rewritten)
#pragma unroll
  for (int i = 0; i < 32; ++i) {
    const int j = lane * 32 + i;           // 0..1023 over (d, key-16)
    vw[(j >> 4) * 32 + 16 + (j & 15)] = f2bf(0.0f);
  }

  const __bf16* Qb = Q + ((size_t)bh * Tt + q0) * HDd;
  const v16bf qf0 = load_frag(Qb, 0, HDd, 0, lane);
  const v16bf qf1 = load_frag(Qb, 0, HDd, 32, lane);

  float mrow[8], lrow[8];
  v8f o[4];
#pragma unroll
  for (int r = 0; r < 8; ++r) { mrow[r] = -3.0e38f; lrow[r] = 0.0f; }
#pragma unroll
  for (int s = 0; s < 4; ++s)
#pragma unroll
    for (int i = 0; i < 8; ++i) o[s][i] = 0.0f;

  for (int k0 = 0; k0 <= q0; k0 += 16) {
    // ---- S = Q K^T (16x16, fp32) ----
    const __bf16* Kb = Km + ((size_t)bh * Tt + k0) * HDd;
    v8f s;
#pragma unroll
    for (int i = 0; i < 8; ++i) s[i] = 0.0f;
    s = wmma_bf16(qf0, load_frag(Kb, 0, HDd, 0, lane), s);
    s = wmma_bf16(qf1, load_frag(Kb, 0, HDd, 32, lane), s);

    // ---- banded rel term: rows [base .. base+31], base in [1, 2R-15] ----
    const int base = k0 - q0 - 15 + Rr;
    const __bf16* Rb = REL + (size_t)base * HDd;
    v8f r0, r1;
#pragma unroll
    for (int i = 0; i < 8; ++i) { r0[i] = 0.0f; r1[i] = 0.0f; }
    r0 = wmma_bf16(qf0, load_frag(Rb, 0, HDd, 0, lane), r0);
    r0 = wmma_bf16(qf1, load_frag(Rb, 0, HDd, 32, lane), r0);
    r1 = wmma_bf16(qf0, load_frag(Rb, 16, HDd, 0, lane), r1);
    r1 = wmma_bf16(qf1, load_frag(Rb, 16, HDd, 32, lane), r1);
#pragma unroll
    for (int r = 0; r < 8; ++r) {
      const int m = r + 8 * hi;
      relw[m * 32 + ln]      = r0[r];
      relw[m * 32 + 16 + ln] = r1[r];
    }
    asm volatile("s_wait_dscnt 0x0" ::: "memory");

    // stage V tile transposed into LDS: vw[hd][key 0..15]
    {
      const __bf16* Vb = Vm + ((size_t)bh * Tt + k0 + ln) * HDd + hi * 32;
#pragma unroll
      for (int j = 0; j < 4; ++j) {
        v8bf vv = *(const v8bf*)(Vb + j * 8);
#pragma unroll
        for (int e = 0; e < 8; ++e) vw[(hi * 32 + j * 8 + e) * 32 + ln] = vv[e];
      }
    }

    const bool pad = (x[b * Tt + k0 + ln] == 0);

    // ---- online softmax (rows live across 16 lanes of each half) ----
    float p8[8], corr[8];
#pragma unroll
    for (int r = 0; r < 8; ++r) {
      const int m = r + 8 * hi;
      float sv = s[r] + relw[m * 32 + (ln - m + 15)];  // gather Toeplitz band
      sv *= 0.125f;                                    // 1/sqrt(HD)
      if ((k0 + ln) > (q0 + m) || pad) sv = -1.0e9f;
      const float tmax = redmax16(sv);
      const float nm = fmaxf(mrow[r], tmax);
      const float p = __expf(sv - nm);
      const float tsum = redsum16(p);
      corr[r] = __expf(mrow[r] - nm);
      lrow[r] = lrow[r] * corr[r] + tsum;
      mrow[r] = nm;
      p8[r] = p;
    }
#pragma unroll
    for (int ss = 0; ss < 4; ++ss)
#pragma unroll
      for (int r = 0; r < 8; ++r) o[ss][r] *= corr[r];

    // P -> LDS, then build 16x32 A fragment (upper K half zero)
#pragma unroll
    for (int r = 0; r < 8; ++r) pw[(r + 8 * hi) * 16 + ln] = f2bf(p8[r]);
    asm volatile("s_wait_dscnt 0x0" ::: "memory");

    v16bf pf;
    {
      v8bf lo = *(const v8bf*)(pw + ln * 16 + hi * 8);
#pragma unroll
      for (int i = 0; i < 8; ++i) { pf[i] = lo[i]; pf[i + 8] = f2bf(0.0f); }
    }
#pragma unroll
    for (int ss = 0; ss < 4; ++ss) {
      v16bf vf = load_frag(vw, ss * 16, 32, 0, lane);
      o[ss] = wmma_bf16(pf, vf, o[ss]);
    }
  }

  // ---- normalize and write O as [B,T,D] bf16 (h-major columns) ----
#pragma unroll
  for (int ss = 0; ss < 4; ++ss)
#pragma unroll
    for (int r = 0; r < 8; ++r) {
      const int m = r + 8 * hi;
      const float ov = o[ss][r] / lrow[r];
      O[((size_t)b * Tt + q0 + m) * Dd + h * HDd + ss * 16 + ln] = f2bf(ov);
    }
}

// ---------------- elementwise kernels ----------------
__global__ __launch_bounds__(256) void embed_kernel(
    const int* __restrict__ x, const float* __restrict__ emb,
    float* __restrict__ hf, __bf16* __restrict__ hb) {
  const int idx = blockIdx.x * 256 + threadIdx.x;   // over BT*D
  const int d = idx & (Dd - 1);
  const int row = idx >> 9;                          // D = 512
  const int t = row & (Tt - 1);
  const int tok = x[row];
  const float freq = __expf((float)(d & ~1) * (-9.210340372f / (float)Dd));
  const float ang = (float)t * freq;
  const float pe = (d & 1) ? __cosf(ang) : __sinf(ang);
  const float v = emb[(size_t)tok * Dd + d] * 22.627416998f + pe;  // sqrt(512)
  hf[idx] = v;
  hb[idx] = f2bf(v);
}

__global__ __launch_bounds__(256) void add_ln_kernel(
    float* __restrict__ hf, const float* __restrict__ tmp,
    const float* __restrict__ sc, const float* __restrict__ bi,
    __bf16* __restrict__ hb) {
  const int row = blockIdx.x;
  const int tid = threadIdx.x;
  const size_t base = (size_t)row * Dd;
  const float a0 = hf[base + tid] + tmp[base + tid];
  const float a1 = hf[base + tid + 256] + tmp[base + tid + 256];
  float sum = a0 + a1, sq = a0 * a0 + a1 * a1;
#pragma unroll
  for (int m = 16; m >= 1; m >>= 1) {
    sum += __shfl_xor(sum, m, 32);
    sq  += __shfl_xor(sq, m, 32);
  }
  __shared__ float ws1[8], ws2[8];
  const int w = tid >> 5, l = tid & 31;
  if (l == 0) { ws1[w] = sum; ws2[w] = sq; }
  __syncthreads();
  if (w == 0) {
    float s1 = (l < 8) ? ws1[l] : 0.0f;
    float s2 = (l < 8) ? ws2[l] : 0.0f;
#pragma unroll
    for (int m = 4; m >= 1; m >>= 1) {
      s1 += __shfl_xor(s1, m, 32);
      s2 += __shfl_xor(s2, m, 32);
    }
    if (l == 0) { ws1[0] = s1; ws2[0] = s2; }
  }
  __syncthreads();
  const float mean = ws1[0] * (1.0f / (float)Dd);
  const float var  = ws2[0] * (1.0f / (float)Dd) - mean * mean;
  const float inv  = rsqrtf(var + 1e-5f);
  const float o0 = (a0 - mean) * inv * sc[tid] + bi[tid];
  const float o1 = (a1 - mean) * inv * sc[tid + 256] + bi[tid + 256];
  hf[base + tid] = o0;       hf[base + tid + 256] = o1;
  hb[base + tid] = f2bf(o0); hb[base + tid + 256] = f2bf(o1);
}

__global__ __launch_bounds__(256) void gather_kernel(
    const __bf16* __restrict__ hb, const int* __restrict__ pos,
    __bf16* __restrict__ g) {
  const int idx = blockIdx.x * 256 + threadIdx.x;   // over Bb*Mc*Dd
  const int d = idx & (Dd - 1);
  const int bm = idx >> 9;
  const int b = bm / Mc;
  int p = pos[bm];
  if (p < 0) p = 0;
  g[idx] = hb[((size_t)b * Tt + p) * Dd + d];
}

__global__ __launch_bounds__(256) void bf16cvt_kernel(
    const float* __restrict__ in, __bf16* __restrict__ out, int n) {
  const int i = blockIdx.x * 256 + threadIdx.x;
  if (i < n) out[i] = f2bf(in[i]);
}

// ---------------- host driver ----------------
extern "C" void kernel_launch(void* const* d_in, const int* in_sizes, int n_in,
                              void* d_out, int out_size, void* d_ws,
                              size_t ws_size, hipStream_t stream) {
  const int*   x    = (const int*)d_in[0];
  const int*   cpos = (const int*)d_in[1];
  const float* emb  = (const float*)d_in[2];
  const float* rel  = (const float*)d_in[3];
  const float* Wq = (const float*)d_in[4];   const float* bq = (const float*)d_in[5];
  const float* Wk = (const float*)d_in[6];   const float* bk = (const float*)d_in[7];
  const float* Wv = (const float*)d_in[8];   const float* bv = (const float*)d_in[9];
  const float* Wo = (const float*)d_in[10];  const float* bo = (const float*)d_in[11];
  const float* l1s = (const float*)d_in[12]; const float* l1b = (const float*)d_in[13];
  const float* W1 = (const float*)d_in[14];  const float* b1 = (const float*)d_in[15];
  const float* W2 = (const float*)d_in[16];  const float* b2 = (const float*)d_in[17];
  const float* l2s = (const float*)d_in[18]; const float* l2b = (const float*)d_in[19];
  const float* Wout = (const float*)d_in[20]; const float* bout = (const float*)d_in[21];
  const float* Wc = (const float*)d_in[22];  const float* bc = (const float*)d_in[23];

  float* out = (float*)d_out;

  char* w = (char*)d_ws;
  auto take = [&](size_t bytes) {
    char* p = w;
    w += (bytes + 255) & ~(size_t)255;
    return p;
  };
  float*  hf   = (float*) take((size_t)BT * Dd * 4);
  float*  tmp  = (float*) take((size_t)BT * Dd * 4);
  __bf16* hb   = (__bf16*)take((size_t)BT * Dd * 2);
  __bf16* qb   = (__bf16*)take((size_t)BT * Dd * 2);
  __bf16* kb   = (__bf16*)take((size_t)BT * Dd * 2);
  __bf16* vb   = (__bf16*)take((size_t)BT * Dd * 2);
  __bf16* ob   = (__bf16*)take((size_t)BT * Dd * 2);
  __bf16* fb   = (__bf16*)take((size_t)BT * Ff * 2);
  __bf16* gb   = (__bf16*)take((size_t)Bb * Mc * Dd * 2);
  __bf16* relb = (__bf16*)take((size_t)Ll * NREL * HDd * 2);

  {  // all layers' rel_emb fp32 -> bf16 once
    const int n = Ll * NREL * HDd;
    bf16cvt_kernel<<<(n + 255) / 256, 256, 0, stream>>>(rel, relb, n);
  }
  embed_kernel<<<(BT * Dd) / 256, 256, 0, stream>>>(x, emb, hf, hb);

  const dim3 gProj(Dd / 64, BT / 64);   // 512-col GEMMs
  const dim3 gFfn1(Ff / 64, BT / 64);
  for (int l = 0; l < Ll; ++l) {
    const float* Wq_l = Wq + (size_t)l * Dd * Dd;
    const float* Wk_l = Wk + (size_t)l * Dd * Dd;
    const float* Wv_l = Wv + (size_t)l * Dd * Dd;
    const float* Wo_l = Wo + (size_t)l * Dd * Dd;
    const float* W1_l = W1 + (size_t)l * Dd * Ff;
    const float* W2_l = W2 + (size_t)l * Ff * Dd;
    const __bf16* rel_l = relb + (size_t)l * NREL * HDd;

    gemm_kernel<OUT_QKV><<<gProj, 128, 0, stream>>>(hb, Wq_l, bq + l * Dd, qb, BT, Dd, Dd);
    gemm_kernel<OUT_QKV><<<gProj, 128, 0, stream>>>(hb, Wk_l, bk + l * Dd, kb, BT, Dd, Dd);
    gemm_kernel<OUT_QKV><<<gProj, 128, 0, stream>>>(hb, Wv_l, bv + l * Dd, vb, BT, Dd, Dd);
    attention_kernel<<<dim3(Tt / 64, Bb * Hh), 128, 0, stream>>>(qb, kb, vb, rel_l, x, ob);
    gemm_kernel<OUT_F32><<<gProj, 128, 0, stream>>>(ob, Wo_l, bo + l * Dd, tmp, BT, Dd, Dd);
    add_ln_kernel<<<BT, 256, 0, stream>>>(hf, tmp, l1s + l * Dd, l1b + l * Dd, hb);
    gemm_kernel<OUT_BF16_RELU><<<gFfn1, 128, 0, stream>>>(hb, W1_l, b1 + l * Ff, fb, BT, Ff, Dd);
    gemm_kernel<OUT_F32><<<gProj, 128, 0, stream>>>(fb, W2_l, b2 + l * Dd, tmp, BT, Dd, Ff);
    add_ln_kernel<<<BT, 256, 0, stream>>>(hf, tmp, l2s + l * Dd, l2b + l * Dd, hb);
  }

  // logits = h @ Wout + bout  -> first BT*V floats of d_out
  gemm_kernel<OUT_F32><<<dim3(Vv / 64, BT / 64), 128, 0, stream>>>(
      hb, Wout, bout, out, BT, Vv, Dd);
  // chord head: gather rows then small GEMM -> after logits
  gather_kernel<<<(Bb * Mc * Dd) / 256, 256, 0, stream>>>(hb, cpos, gb);
  gemm_kernel<OUT_F32><<<dim3(Cc / 64, (Bb * Mc) / 64), 128, 0, stream>>>(
      gb, Wc, bc, out + (size_t)BT * Vv, Bb * Mc, Cc, Dd);
}
